// FPSANConv2d_48112223649986
// MI455X (gfx1250) — matrix operand order repaired and checked by reading, not compiled, not run on hardware
//
#include <hip/hip_runtime.h>
#include <hip/hip_bf16.h>

// ---------------------------------------------------------------------------
// Fused ternary-quantized dual-conv (BitNet-style growth conv) for gfx1250.
// out = conv(x, Qt(w)) + b + sigmoid(gate) * (conv(x, Qt(w_child)) + b_child)
// Conv is linear in W  =>  single conv with W_eff = Qt(w) + g*Qt(w_child).
// Implicit GEMM:  D[64, N*H*W] = W_eff[64,576] x im2col(x)[576, N*H*W]
// using v_wmma_f32_16x16x32_bf16 (wave32) + async global->LDS for the
// weight-tile fill.
// ---------------------------------------------------------------------------

typedef __attribute__((ext_vector_type(16))) __bf16        v16bf;
typedef __attribute__((ext_vector_type(8)))  float         v8f;
typedef __attribute__((ext_vector_type(4)))  unsigned int  v4u;
typedef __attribute__((ext_vector_type(8)))  unsigned int  v8u;
typedef __attribute__((ext_vector_type(4)))  int           v4i;

#define KTOT      576      // C_in(64) * 3 * 3
#define KSTEPS    18       // 576 / 32
#define STRIDE_A  584      // padded LDS row stride (elems) for W_eff  (mult of 8)
#define STRIDE_B  40       // padded LDS row stride (elems) for im2col (mult of 8)
#define A_BYTES   (64 * STRIDE_A * 2)          // 74752
#define B_ELEMS   (128 * STRIDE_B)             // per buffer
#define B_BYTES   (B_ELEMS * 2)                // 10240 per buffer
#define SMEM_BYTES (A_BYTES + 2 * B_BYTES)     // 95232

#if defined(__gfx1250__) && __has_builtin(__builtin_amdgcn_global_load_async_to_lds_b128)
#define USE_ASYNC_LDS 1
#else
#define USE_ASYNC_LDS 0
#endif

// Pointer flavors matching the async builtin signature:
//   param0: v4i in global (__device__, AS1), param1: v4i in LDS (AS3).
typedef __attribute__((address_space(1))) v4i gv4i_t;
typedef __attribute__((address_space(3))) v4i lv4i_t;

__device__ __forceinline__ gv4i_t* to_gbl_v4i(const void* p) {
  return (gv4i_t*)(uintptr_t)p;
}
__device__ __forceinline__ lv4i_t* to_lds_v4i(void* p) {
  // ISA: flat addresses in the LDS aperture map to LDS via addr[31:0].
  return (lv4i_t*)(unsigned)(uintptr_t)p;
}

__device__ __forceinline__ unsigned int f2bf_bits(float f) {
  unsigned int u = __builtin_bit_cast(unsigned int, f);
  return (u + 0x7fffu + ((u >> 16) & 1u)) >> 16;   // round-to-nearest-even
}

// Build a v16bf fragment from two 16-byte LDS chunks.
__device__ __forceinline__ v16bf load_frag2(const __bf16* p, int elemOff1) {
  v4u a = *(const v4u*)p;
  v4u b = *(const v4u*)(p + elemOff1);
  v8u r;
  r[0] = a[0]; r[1] = a[1]; r[2] = a[2]; r[3] = a[3];
  r[4] = b[0]; r[5] = b[1]; r[6] = b[2]; r[7] = b[3];
  return __builtin_bit_cast(v16bf, r);
}

// ---------------------------------------------------------------------------
// Kernel 1: scale = max(mean(|w|), 1e-5) for w (block 0) and w_child (block 1)
// ---------------------------------------------------------------------------
__global__ __launch_bounds__(256) void scales_kernel(const float* __restrict__ w,
                                                     const float* __restrict__ wc,
                                                     float* __restrict__ wsf, int n) {
  const float* src = (blockIdx.x == 0) ? w : wc;
  int tid = threadIdx.x;
  float s = 0.f;
  for (int i = tid; i < n; i += 256) s += fabsf(src[i]);
  __shared__ float red[256];
  red[tid] = s;
  __syncthreads();
  for (int o = 128; o > 0; o >>= 1) {
    if (tid < o) red[tid] += red[tid + o];
    __syncthreads();
  }
  if (tid == 0) wsf[blockIdx.x] = fmaxf(red[0] / (float)n, 1e-5f);
}

// ---------------------------------------------------------------------------
// Kernel 2: W_eff (bf16, K-permuted) = Qt(w) + g*Qt(w_child); b_eff = b+g*bc
// dst index: o*576 + (ky*3+kx)*64 + c   from src idx = o*576 + c*9 + ky*3 + kx
// ---------------------------------------------------------------------------
__global__ __launch_bounds__(256) void build_kernel(const float* __restrict__ w,
                                                    const float* __restrict__ wc,
                                                    const float* __restrict__ b,
                                                    const float* __restrict__ bc,
                                                    const float* __restrict__ gate,
                                                    const float* __restrict__ wsf,
                                                    float* __restrict__ beff,
                                                    unsigned short* __restrict__ weff,
                                                    int n) {
  int idx = blockIdx.x * 256 + threadIdx.x;
  float g = 1.f / (1.f + expf(-gate[0]));
  if (idx < n) {
    float s0 = wsf[0], s1 = wsf[1];
    float q0 = fminf(fmaxf(rintf(w[idx]  / s0), -1.f), 1.f) * s0;
    float q1 = fminf(fmaxf(rintf(wc[idx] / s1), -1.f), 1.f) * s1;
    unsigned o   = (unsigned)idx / 576u;
    unsigned rem = (unsigned)idx - o * 576u;
    unsigned c   = rem / 9u;
    unsigned r9  = rem - c * 9u;                 // ky*3+kx
    weff[o * 576u + r9 * 64u + c] = (unsigned short)f2bf_bits(q0 + g * q1);
  }
  if (blockIdx.x == 0 && threadIdx.x < 64)
    beff[threadIdx.x] = b[threadIdx.x] + g * bc[threadIdx.x];
}

// ---------------------------------------------------------------------------
// Stage one 32-wide K-chunk of im2col for this thread: 16 contiguous channels
// at a fixed conv tap (ky,kx). One base address + immediate offsets.
// ---------------------------------------------------------------------------
__device__ __forceinline__ void stage_chunk(int j, int row, int px, int krow,
                                            const float* __restrict__ xbase,
                                            float* stage) {
  int pos = j >> 1;                    // 0..8  (scalar)
  int ky  = pos / 3;                   // scalar div
  int kx  = pos - ky * 3;
  int ih  = row + ky - 1;
  int iw  = px + kx - 1;
  bool vrow = (unsigned)ih < 128u;
  bool vcol = (unsigned)iw < 128u;
  int ihc = vrow ? ih : 0;
  int iwc = min(max(iw, 0), 127);
  bool valid = vrow && vcol;
  int c0 = (j & 1) * 32 + krow * 16;   // first channel this thread stages
  const float* p = xbase + ((size_t)c0 * 128 + ihc) * 128 + iwc;
#pragma unroll
  for (int i = 0; i < 16; ++i) {
    float v = p[(size_t)i * 16384];    // channel stride: 64KB immediate
    stage[i] = valid ? v : 0.f;
  }
}

__device__ __forceinline__ void commit_chunk(unsigned int* __restrict__ dst,
                                             const float* stage) {
#pragma unroll
  for (int m = 0; m < 8; ++m) {
    unsigned int u = f2bf_bits(stage[2 * m]) | (f2bf_bits(stage[2 * m + 1]) << 16);
    dst[m] = u;                         // packed 2x bf16, ds_store_b32
  }
}

// ---------------------------------------------------------------------------
// Kernel 3: implicit-GEMM conv with WMMA.
// Grid: 2048 blocks = 16 images * 128 output rows. Block: 256 thr = 8 waves.
// ---------------------------------------------------------------------------
__global__ __launch_bounds__(256) void conv_wmma_kernel(const float* __restrict__ x,
                                                        const unsigned short* __restrict__ weff,
                                                        const float* __restrict__ beff,
                                                        float* __restrict__ out) {
  extern __shared__ char smem[];
  __bf16* Alds = (__bf16*)smem;                 // [64][STRIDE_A]
  __bf16* Blds = (__bf16*)(smem + A_BYTES);     // [2][128][STRIDE_B]

  const int tid = threadIdx.x;
  const int img = blockIdx.x >> 7;
  const int row = blockIdx.x & 127;

  // ---- fill of full W_eff into LDS (padded rows): async DMA if available ----
  {
    const uint4* src = (const uint4*)weff;      // 4608 16-byte chunks
#pragma unroll
    for (int i = 0; i < 18; ++i) {
      int chunk = tid + i * 256;
      int r  = chunk / 72;                      // 72 chunks per 576-elem row
      int c8 = chunk - r * 72;
      __bf16* dst = Alds + r * STRIDE_A + c8 * 8;
#if USE_ASYNC_LDS
      __builtin_amdgcn_global_load_async_to_lds_b128(to_gbl_v4i(src + chunk),
                                                     to_lds_v4i(dst), 0, 0);
#else
      *(uint4*)dst = src[chunk];
#endif
    }
  }

  const int px   = tid & 127;                   // output column this thread feeds
  const int krow = tid >> 7;                    // 0/1: which 16 k-locals it stages
  const float* xbase = x + (size_t)img * 64 * 128 * 128;

  // LDS store base for this thread's 16 packed k-locals
  unsigned int* bStore =
      (unsigned int*)((unsigned short*)Blds + px * STRIDE_B + krow * 16);

  // ---- prologue: chunk 0 ----
  {
    float stage[16];
    stage_chunk(0, row, px, krow, xbase, stage);
    commit_chunk(bStore, stage);
  }
#if USE_ASYNC_LDS
#if __has_builtin(__builtin_amdgcn_s_wait_asynccnt)
  __builtin_amdgcn_s_wait_asynccnt(0);
#else
  asm volatile("s_wait_asynccnt 0" ::: "memory");
#endif
#endif
  __syncthreads();

  // ---- per-wave fragment addressing (contiguous k = 16*(lane/16) + s in both A and B) ----
  const int wave   = tid >> 5;
  const int lane   = tid & 31;
  const int mstrip = wave >> 1;                 // 0..3 -> C_out rows 16*mstrip..+15
  const int nhalf  = wave & 1;                  // 0/1  -> px 0..63 / 64..127
  const int aRow   = mstrip * 16 + (lane & 15);
  const int aK     = (lane < 16) ? 0 : 16;     // A: lanes0-15 K0..15, 16-31 K16..31
  const int bCol   = nhalf * 64 + (lane & 15);
  const int bK     = (lane >> 4) << 4;         // B: lanes0-15 K0..15, 16-31 K16..31

  const __bf16* aPtr    = Alds + aRow * STRIDE_A + aK;
  const int     bColOff = bCol * STRIDE_B + bK;

  v8f acc[4] = {};

  for (int kc = 0; kc < KSTEPS; ++kc) {
    // stage next im2col chunk (global loads issued early)
    float stage[16];
    const bool has_next = (kc + 1) < KSTEPS;
    if (has_next) stage_chunk(kc + 1, row, px, krow, xbase, stage);

    // compute on current buffer
    const __bf16* Bcur = Blds + (kc & 1) * B_ELEMS;
    v16bf afrag = load_frag2(aPtr + kc * 32, 8);
#pragma unroll
    for (int t = 0; t < 4; ++t) {
      v16bf bfrag = load_frag2(Bcur + bColOff + t * 16 * STRIDE_B, 8);
      acc[t] = __builtin_amdgcn_wmma_f32_16x16x32_bf16(
          false, afrag, false, bfrag, (short)0, acc[t], false, false);
    }

    // commit staged chunk into the other buffer
    if (has_next)
      commit_chunk(bStore + ((kc + 1) & 1) * (B_ELEMS / 2), stage);
    __syncthreads();
  }

  // ---- epilogue: C/D layout (VGPR i: lanes0-15 M=i, lanes16-31 M=i+8) ----
  const int nlane = lane & 15;
  const int mAdd  = (lane < 16) ? 0 : 8;
#pragma unroll
  for (int t = 0; t < 4; ++t) {
#pragma unroll
    for (int i = 0; i < 8; ++i) {
      int cout = mstrip * 16 + mAdd + i;
      int ww   = nhalf * 64 + t * 16 + nlane;
      out[(((size_t)img * 64 + cout) * 128 + row) * 128 + ww] =
          acc[t][i] + beff[cout];
    }
  }
}

// ---------------------------------------------------------------------------
extern "C" void kernel_launch(void* const* d_in, const int* in_sizes, int n_in,
                              void* d_out, int out_size, void* d_ws, size_t ws_size,
                              hipStream_t stream) {
  const float* x    = (const float*)d_in[0];
  const float* w    = (const float*)d_in[1];
  const float* b    = (const float*)d_in[2];
  const float* wc   = (const float*)d_in[3];
  const float* bc   = (const float*)d_in[4];
  const float* gate = (const float*)d_in[5];
  float* out = (float*)d_out;

  float* wsf  = (float*)d_ws;                       // [0..1] scales
  float* beff = wsf + 8;                            // [64] fused bias
  unsigned short* weffbf = (unsigned short*)((char*)d_ws + 512);  // 36864 bf16

  const int n = 64 * 64 * 9;                        // 36864

  scales_kernel<<<2, 256, 0, stream>>>(w, wc, wsf, n);
  build_kernel<<<(n + 255) / 256, 256, 0, stream>>>(w, wc, b, bc, gate, wsf,
                                                    beff, weffbf, n);
  conv_wmma_kernel<<<16 * 128, 256, SMEM_BYTES, stream>>>(x, weffbf, beff, out);
}
